// FactorizationMachine_90134183673918
// MI455X (gfx1250) — matrix operand order repaired and checked by reading, not compiled, not run on hardware
//
#include <hip/hip_runtime.h>

// Factorization Machine on MI455X (gfx1250).
//
// out[b] = x_b.W + b0 + 0.5*( ||x_b V||^2 - (x_b*x_b).r ),  r_d = sum_k V[d,k]^2
//
// Roofline: streaming x once = 128 MB fp32 @ 23.3 TB/s ~= 5.5 us floor.
// x@V (8.6 GFLOP) needs >1.6 PFLOPS to stay memory-bound => bf16 WMMA with
// bf16x3 (hi/lo split) fp32 emulation: 3x WMMA work, still far below the
// bf16 matrix ceiling, ~fp32 accuracy.
//
// V is pre-converted ONCE into d_ws as the exact padded/transposed LDS tile
// image (hi+lo bf16, [n][40] rows), so the main loop stages V with pure
// 16B async-to-LDS copies (ASYNCcnt), double-buffered to overlap with the
// WMMA stream. Only x is converted on the fly (it streams once).

#define INPUT_DIM 2048
#define EMBED_DIM 128
#define BATCH     16384
#define KS        40            // padded LDS row stride (bf16): 80B, 16B-aligned, conflict-free
#define TILE_HALF (EMBED_DIM * KS)        // 5120 ushorts  (10240 B): one hi or lo block
#define TILE_US   (2 * TILE_HALF)         // 10240 ushorts (20480 B): hi + lo per chunk
#define NCHUNK    (INPUT_DIM / 32)        // 64

typedef __attribute__((ext_vector_type(16))) __bf16 v16bf;
typedef __attribute__((ext_vector_type(8)))  float  v8f;

__device__ __forceinline__ unsigned short bf_bits(__bf16 h) {
  union { __bf16 b; unsigned short u; } c; c.b = h; return c.u;
}

// ---------------------------------------------------------------------------
// async global->LDS copy (gfx1250 ASYNCcnt path) with sync fallback
// ---------------------------------------------------------------------------
#if defined(__has_builtin)
#if __has_builtin(__builtin_amdgcn_global_load_async_to_lds_b128)
#define FM_ASYNC_CP 1
#endif
#endif

#ifdef FM_ASYNC_CP
#pragma message("FM: using global_load_async_to_lds_b128 path")
// The builtin wants native 16B int vectors: (v4i AS1* src, v4i AS3* dst, Ii, Ii)
typedef int v4i __attribute__((vector_size(16)));
typedef __attribute__((address_space(1))) v4i g_v4i;
typedef __attribute__((address_space(3))) v4i l_v4i;
#else
#pragma message("FM: async-to-LDS builtin unavailable, sync uint4 copy fallback")
#endif

__device__ __forceinline__ void cp16(uint4* dst_lds, const uint4* src_glb) {
#ifdef FM_ASYNC_CP
  __builtin_amdgcn_global_load_async_to_lds_b128((g_v4i*)src_glb,
                                                 (l_v4i*)dst_lds, 0, 0);
#else
  *dst_lds = *src_glb;
#endif
}

__device__ __forceinline__ void async_drain() {
#ifdef FM_ASYNC_CP
  asm volatile("s_wait_asynccnt 0x0" ::: "memory");
#endif
}

// ---------------------------------------------------------------------------
// prep 1: r[d] = sum_k V[d,k]^2
// ---------------------------------------------------------------------------
__global__ void fm_prep_r(const float* __restrict__ V, float* __restrict__ r) {
  int d = blockIdx.x * blockDim.x + threadIdx.x;
  if (d < INPUT_DIM) {
    const float* row = V + (size_t)d * EMBED_DIM;
    float s = 0.f;
#pragma unroll 8
    for (int k = 0; k < EMBED_DIM; ++k) { float v = row[k]; s = fmaf(v, v, s); }
    r[d] = s;
  }
}

// ---------------------------------------------------------------------------
// prep 2: build per-chunk LDS tile images in ws:
//   chunk c (K rows c*32..c*32+31), transposed [n][k], padded stride KS:
//   hi at tiles[c*TILE_US + n*KS + k], lo at tiles[c*TILE_US + TILE_HALF + ...]
// ---------------------------------------------------------------------------
__global__ void fm_prep_tile(const float* __restrict__ V,
                             unsigned short* __restrict__ tiles) {
  int e = blockIdx.x * blockDim.x + threadIdx.x;   // 0 .. 64*32*128-1
  int c = e >> 12;                                  // chunk (4096 elems each)
  int f = e & 4095;
  int k = f >> 7;                                   // 0..31
  int n = f & 127;                                  // coalesced read over n
  float v = V[(size_t)(c * 32 + k) * EMBED_DIM + n];
  __bf16 h = (__bf16)v;
  __bf16 l = (__bf16)(v - (float)h);
  size_t base = (size_t)c * TILE_US + (size_t)n * KS + k;
  tiles[base]             = bf_bits(h);
  tiles[base + TILE_HALF] = bf_bits(l);
}

// ---------------------------------------------------------------------------
// main: 128 WGs x 256 thr (8 waves); each wave owns a 16-row strip.
// Double-buffered LDS, per-chunk pipeline:
//   issue copy(next) -> compute(cur): A-convert + dots + 24 WMMA -> drain -> barrier
// ---------------------------------------------------------------------------
__global__ __launch_bounds__(256) void fm_main(const float* __restrict__ x,
                                               const float* __restrict__ W,
                                               const float* __restrict__ bias,
                                               const unsigned short* __restrict__ tiles,
                                               const float* __restrict__ r,
                                               float* __restrict__ out) {
  __shared__ unsigned short Vt[2][TILE_US];   // [buf][hi(5120) | lo(5120)] = 2 x 20 KB

  const int tid  = threadIdx.x;
  const int lane = tid & 31;
  const int wave = tid >> 5;
  const int half = lane >> 4;      // 0: lanes 0-15, 1: lanes 16-31
  const int mrow = lane & 15;      // row within the wave's 16-row strip

  const int rowbase = blockIdx.x * 128 + wave * 16;
  const float* xr = x + (size_t)(rowbase + mrow) * INPUT_DIM;

  v8f acc[8] = {};                 // 8 N-tiles of 16x16 f32 accumulators
  float lin = 0.f, sqr = 0.f;

  // ---- preload chunk 0 into buf 0 ----
  {
    const uint4* g = (const uint4*)tiles;       // chunk 0 image
    uint4* l = (uint4*)&Vt[0][0];
#pragma unroll
    for (int i = 0; i < 5; ++i) cp16(l + tid + i * 256, g + tid + i * 256);
  }
  async_drain();
  __syncthreads();

  int cur = 0;
  for (int c = 0; c < NCHUNK; ++c) {
    const int kc = c * 32;

    // ---- issue copy of next chunk into the other buffer ----
    if (c + 1 < NCHUNK) {
      const uint4* g = (const uint4*)(tiles + (size_t)(c + 1) * TILE_US);
      uint4* l = (uint4*)&Vt[cur ^ 1][0];
#pragma unroll
      for (int i = 0; i < 5; ++i) cp16(l + tid + i * 256, g + tid + i * 256);
    }

    // ---- A fragment: x[row][kc..kc+31] in WMMA bf16 A layout ----
    // lanes 0-15: K 0-7 & 16-23 ; lanes 16-31: K 8-15 & 24-31
    const int c0 = kc + half * 8;
    const int c1 = kc + 16 + half * 8;
    float xs[16];
    *(float4*)&xs[0]  = *(const float4*)(xr + c0);
    *(float4*)&xs[4]  = *(const float4*)(xr + c0 + 4);
    *(float4*)&xs[8]  = *(const float4*)(xr + c1);
    *(float4*)&xs[12] = *(const float4*)(xr + c1 + 4);

    if (kc + 128 < INPUT_DIM) __builtin_prefetch(xr + kc + 128, 0, 0);

    v16bf ahi, alo;
#pragma unroll
    for (int i = 0; i < 16; ++i) {
      __bf16 h = (__bf16)xs[i];          // native RNE convert (v_cvt class)
      ahi[i] = h;
      alo[i] = (__bf16)(xs[i] - (float)h);
    }

    // ---- exact fp32 first-order + squared-norm dots ----
#pragma unroll
    for (int i = 0; i < 8; ++i) {
      lin = fmaf(xs[i],                 W[c0 + i], lin);
      sqr = fmaf(xs[i] * xs[i],         r[c0 + i], sqr);
      lin = fmaf(xs[8 + i],             W[c1 + i], lin);
      sqr = fmaf(xs[8 + i] * xs[8 + i], r[c1 + i], sqr);
    }

    // ---- 8 N-tiles, bf16x3 fp32-emulated WMMA ----
    const unsigned short* hbase = &Vt[cur][0];
#pragma unroll
    for (int t = 0; t < 8; ++t) {
      const unsigned short* ph = hbase + (t * 16 + mrow) * KS + half * 8;
      const unsigned short* pl = ph + TILE_HALF;
      union BF { uint4 q[2]; v16bf v; } bhi, blo;
      bhi.q[0] = *(const uint4*)ph;  bhi.q[1] = *(const uint4*)(ph + 16);
      blo.q[0] = *(const uint4*)pl;  blo.q[1] = *(const uint4*)(pl + 16);
      acc[t] = __builtin_amdgcn_wmma_f32_16x16x32_bf16(
          false, ahi, false, bhi.v, (short)0, acc[t], false, false);
      acc[t] = __builtin_amdgcn_wmma_f32_16x16x32_bf16(
          false, ahi, false, blo.v, (short)0, acc[t], false, false);
      acc[t] = __builtin_amdgcn_wmma_f32_16x16x32_bf16(
          false, alo, false, bhi.v, (short)0, acc[t], false, false);
    }

    async_drain();       // this wave's next-chunk copies have landed in LDS
    __syncthreads();     // all waves done reading cur + done writing next
    cur ^= 1;
  }

  // ---- epilogue -----------------------------------------------------------
  // combine K-halves of the fp32 dots: lane L and L+16 hold the same row
  lin += __shfl_xor(lin, 16, 32);
  sqr += __shfl_xor(sqr, 16, 32);
  float Aval = lin + bias[0] - 0.5f * sqr;    // per row = lane&15

  // ||xv||^2 per row: C layout VGPR j -> row j (lanes 0-15) / row 8+j (16-31)
  float ssj[8];
#pragma unroll
  for (int j = 0; j < 8; ++j) {
    float s = 0.f;
#pragma unroll
    for (int t = 0; t < 8; ++t) { float cc = acc[t][j]; s = fmaf(cc, cc, s); }
#pragma unroll
    for (int m = 1; m <= 8; m <<= 1) s += __shfl_xor(s, m, 32);
    ssj[j] = s;    // all lanes of each half hold the half's row total
  }

#pragma unroll
  for (int j = 0; j < 8; ++j) {
    float a0 = __shfl(Aval, j, 32);        // row j
    float a1 = __shfl(Aval, 8 + j, 32);    // row 8+j
    if (lane == 0)  out[rowbase + j]      = a0 + 0.5f * ssj[j];
    if (lane == 16) out[rowbase + 8 + j]  = a1 + 0.5f * ssj[j];
  }
}

// ---- launch ---------------------------------------------------------------
extern "C" void kernel_launch(void* const* d_in, const int* in_sizes, int n_in,
                              void* d_out, int out_size, void* d_ws, size_t ws_size,
                              hipStream_t stream) {
  const float* x = (const float*)d_in[0];   // [16384, 2048]
  const float* W = (const float*)d_in[1];   // [1, 2048]
  const float* b = (const float*)d_in[2];   // [1]
  const float* V = (const float*)d_in[3];   // [2048, 128]
  float* out = (float*)d_out;               // [16384]

  // ws layout: r (2048 f32 = 8 KB) | tile images (64 * 20480 B = 1.25 MB)
  float* r = (float*)d_ws;
  unsigned short* tiles = (unsigned short*)((char*)d_ws + 8192);

  fm_prep_r<<<(INPUT_DIM + 255) / 256, 256, 0, stream>>>(V, r);
  fm_prep_tile<<<(NCHUNK * 32 * EMBED_DIM) / 256, 256, 0, stream>>>(V, tiles);
  fm_main<<<BATCH / 128, 256, 0, stream>>>(x, W, b, tiles, r, out);
}
